// GNNLayer_15496242004305
// MI455X (gfx1250) — compile-verified, matrix operand order checked by reading, not run on hardware
//
#include <hip/hip_runtime.h>
#include <hip/hip_bf16.h>
#include <math.h>

typedef float  v8f  __attribute__((ext_vector_type(8)));
typedef float  v4f  __attribute__((ext_vector_type(4)));
typedef __bf16 v16bf __attribute__((ext_vector_type(16)));
typedef unsigned int v4u __attribute__((ext_vector_type(4)));
typedef unsigned int v2u __attribute__((ext_vector_type(2)));

#define B_    8
#define N_    256
#define C_    128
#define NODES (B_ * N_)          // 2048
#define JS    4                  // j-range splits per (b, iTile)
#define JC    (N_ / JS)          // 64 j per block
#define CH    8                  // j's staged per LDS chunk
#define LDS_PITCH (C_ + 8)       // halves, padded against bank conflicts

// ---- workspace layout (bytes) ----
#define WFOLD_OFF  0u            // ushort[128*128]        bf16 W_edge@W_pre2
#define BFOLD_OFF  32768u        // float[128]
#define PI_OFF     33280u        // float[2048*128]
#define PJ_OFF     1081856u      // float[2048*128]
#define WPL_OFF    2130432u      // float[640*128]         W_post@W_lin
#define BPL_OFF    2458112u      // float[128]
#define PART_OFF   2458624u      // float[2048*4*JS*128]   partial stats
#define AGG_OFF    19235840u     // float[2048*4*128]      mean/min/max/std
#define OUT2_OFF   23430144u     // float[2048*128]        pre-GraphNorm
#define MUA_OFF    24478720u     // float[128]             alpha*mu
#define SCL_OFF    24479232u     // float[128]             gn_w * rstd

__device__ __forceinline__ unsigned short f2bf(float f) {
  unsigned u = __float_as_uint(f);
  unsigned r = u + 0x7FFFu + ((u >> 16) & 1u);   // round-to-nearest-even
  return (unsigned short)(r >> 16);
}

// ---------- phase 0: constant folding ----------
__global__ void k_fold_edge(const float* __restrict__ We, const float* __restrict__ be,
                            const float* __restrict__ Wpre, const float* __restrict__ bpre,
                            unsigned short* __restrict__ wfold, float* __restrict__ bfold) {
  int t = blockIdx.x * 256 + threadIdx.x;
  if (t < C_ * C_) {
    int k = t / C_, n = t % C_;
    float s = 0.f;
    for (int c = 0; c < C_; ++c) s += We[k * C_ + c] * Wpre[(2 * C_ + c) * C_ + n];
    wfold[k * C_ + n] = f2bf(s);
  } else if (t < C_ * C_ + C_) {
    int n = t - C_ * C_;
    float s = bpre[n];
    for (int c = 0; c < C_; ++c) s += be[c] * Wpre[(2 * C_ + c) * C_ + n];
    bfold[n] = s;
  }
}

__global__ void k_pre_xproj(const float* __restrict__ x, const float* __restrict__ Wpre,
                            float* __restrict__ Pi, float* __restrict__ Pj) {
  int t = blockIdx.x * 256 + threadIdx.x;
  if (t >= 2 * NODES * C_) return;
  int which = t / (NODES * C_);
  int r = t % (NODES * C_);
  int node = r / C_, c = r % C_;
  const float* xr = x + node * C_;
  const float* w  = Wpre + which * C_ * C_;
  float s = 0.f;
  for (int k = 0; k < C_; ++k) s += xr[k] * w[k * C_ + c];
  (which ? Pj : Pi)[node * C_ + c] = s;
}

__global__ void k_fold_post(const float* __restrict__ Wpost, const float* __restrict__ bpost,
                            const float* __restrict__ Wlin, const float* __restrict__ blin,
                            float* __restrict__ Wpl, float* __restrict__ bpl) {
  int t = blockIdx.x * 256 + threadIdx.x;
  if (t < 5 * C_ * C_) {
    int k = t / C_, c = t % C_;
    float s = 0.f;
    for (int u = 0; u < C_; ++u) s += Wpost[k * C_ + u] * Wlin[u * C_ + c];
    Wpl[k * C_ + c] = s;
  } else if (t < 5 * C_ * C_ + C_) {
    int c = t - 5 * C_ * C_;
    float s = blin[c];
    for (int u = 0; u < C_; ++u) s += bpost[u] * Wlin[u * C_ + c];
    bpl[c] = s;
  }
}

// ---------- phase 1: fused edge GEMM + PNA aggregation (WMMA bf16) ----------
__global__ void __launch_bounds__(256, 1)
k_edge_pna(const float* __restrict__ edge, const unsigned short* __restrict__ wfold,
           const float* __restrict__ bfold, const float* __restrict__ Pi,
           const float* __restrict__ Pj, float* __restrict__ part) {
  __shared__ __align__(16) unsigned short a_lds[CH * 16 * LDS_PITCH];

  int bid = blockIdx.x;
  int chunkId = bid % JS;
  int iT = (bid / JS) % (N_ / 16);
  int bb = bid / (JS * (N_ / 16));
  int i0 = iT * 16;

  int tid  = threadIdx.x;
  int wv   = tid >> 5;
  int lane = tid & 31;
  int lhi  = lane >> 4;      // lane-group: selects M 0..7 vs 8..15, K-base 0 vs 8
  int ln   = lane & 15;
  int bse  = lhi * 8;
  int col  = wv * 16 + ln;   // this wave's output column

  // B fragments: W_fold columns for this wave, resident in VGPRs (4 K-steps)
  v16bf Bf[4];
  for (int t = 0; t < 4; ++t) {
    union { unsigned short s[16]; v16bf v; } u;
    #pragma unroll
    for (int h = 0; h < 16; ++h) {
      int k = t * 32 + ((h < 8) ? (bse + h) : (16 + bse + (h - 8)));
      u.s[h] = wfold[k * C_ + col];
    }
    Bf[t] = u.v;
  }

  float bvec[8];
  #pragma unroll
  for (int r = 0; r < 8; ++r) {
    int i = i0 + r + lhi * 8;
    bvec[r] = Pi[(bb * N_ + i) * C_ + col] + bfold[col];
  }

  v8f sum = {0,0,0,0,0,0,0,0};
  v8f sq  = {0,0,0,0,0,0,0,0};
  v8f mn, mx;
  #pragma unroll
  for (int r = 0; r < 8; ++r) { mn[r] = 3.0e38f; mx[r] = -3.0e38f; }

  const float* PjBase = Pj + (size_t)bb * N_ * C_ + col;

  for (int ch = 0; ch < JC / CH; ++ch) {
    int jBase = chunkId * JC + ch * CH;

    // prefetch this chunk's P_j[j, col] values; the loads overlap LDS staging
    float pjv[CH];
    #pragma unroll
    for (int jj = 0; jj < CH; ++jj)
      pjv[jj] = PjBase[(size_t)(jBase + jj) * C_];

    __syncthreads();
    {   // stage 16 i x 8 j x 128 c fp32 tile -> bf16 in LDS
      int r    = tid & 127;            // r = jj*16 + iLocal
      int jj   = r >> 4;
      int il   = r & 15;
      int half = tid >> 7;             // low/high 64 channels
      const float* src = edge +
          ((size_t)(bb * N_ + (i0 + il)) * N_ + (size_t)(jBase + jj)) * C_ + half * 64;
      unsigned short* dst = &a_lds[r * LDS_PITCH + half * 64];
      #pragma unroll
      for (int q = 0; q < 16; ++q) {
        v4f f = *(const v4f*)(src + q * 4);
        v2u p;
        p.x = (unsigned)f2bf(f.x) | ((unsigned)f2bf(f.y) << 16);
        p.y = (unsigned)f2bf(f.z) | ((unsigned)f2bf(f.w) << 16);
        *(v2u*)(dst + q * 4) = p;
      }
    }
    __syncthreads();

    #pragma unroll
    for (int jj = 0; jj < CH; ++jj) {
      v8f acc = {0,0,0,0,0,0,0,0};
      const unsigned short* arow = &a_lds[(jj * 16 + ln) * LDS_PITCH];
      #pragma unroll
      for (int t = 0; t < 4; ++t) {
        union { v4u q[2]; v16bf v; } af;
        af.q[0] = *(const v4u*)(arow + t * 32 + bse);
        af.q[1] = *(const v4u*)(arow + t * 32 + bse + 16);
        acc = __builtin_amdgcn_wmma_f32_16x16x32_bf16(
                  false, af.v, false, Bf[t], (short)0, acc, false, false);
      }
      float pj = pjv[jj];
      #pragma unroll
      for (int r = 0; r < 8; ++r) {
        float v = acc[r] + bvec[r] + pj;          // message m[b, i0+m, j, col]
        sum[r] += v;
        sq[r]   = fmaf(v, v, sq[r]);
        mn[r]   = fminf(mn[r], v);
        mx[r]   = fmaxf(mx[r], v);
      }
    }
  }

  // partial stats: part[node][stat][chunk][c]
  #pragma unroll
  for (int r = 0; r < 8; ++r) {
    int node = bb * N_ + i0 + r + lhi * 8;
    size_t pb = (((size_t)node * 4 + 0) * JS + chunkId) * C_ + col;
    size_t ss = (size_t)JS * C_;
    part[pb]           = sum[r];
    part[pb + ss]      = sq[r];
    part[pb + 2 * ss]  = mn[r];
    part[pb + 3 * ss]  = mx[r];
  }
}

// ---------- phase 2: combine partials -> PNA aggregators ----------
__global__ void k_combine(const float* __restrict__ part, float* __restrict__ agg) {
  int t = blockIdx.x * 256 + threadIdx.x;
  if (t >= NODES * C_) return;
  int node = t / C_, c = t % C_;
  float s = 0.f, q = 0.f, mn = 3.0e38f, mx = -3.0e38f;
  for (int k = 0; k < JS; ++k) {
    s  += part[(((size_t)node * 4 + 0) * JS + k) * C_ + c];
    q  += part[(((size_t)node * 4 + 1) * JS + k) * C_ + c];
    mn  = fminf(mn, part[(((size_t)node * 4 + 2) * JS + k) * C_ + c]);
    mx  = fmaxf(mx, part[(((size_t)node * 4 + 3) * JS + k) * C_ + c]);
  }
  float mean = s * (1.f / N_);
  float var  = q * (1.f / N_) - mean * mean;
  float sd   = sqrtf(fmaxf(var, 0.f) + 1e-5f);
  agg[((size_t)node * 4 + 0) * C_ + c] = mean;
  agg[((size_t)node * 4 + 1) * C_ + c] = mn;
  agg[((size_t)node * 4 + 2) * C_ + c] = mx;
  agg[((size_t)node * 4 + 3) * C_ + c] = sd;
}

// ---------- phase 3: folded post-MLP + final linear ----------
__global__ void k_post(const float* __restrict__ x, const float* __restrict__ agg,
                       const float* __restrict__ Wpl, const float* __restrict__ bpl,
                       float* __restrict__ out2) {
  int t = blockIdx.x * 256 + threadIdx.x;
  if (t >= NODES * C_) return;
  int node = t / C_, c = t % C_;
  float s = bpl[c];
  const float* xr = x + node * C_;
  for (int k = 0; k < C_; ++k) s += xr[k] * Wpl[k * C_ + c];
  const float* ar = agg + (size_t)node * 4 * C_;
  for (int k = 0; k < 4 * C_; ++k) s += ar[k] * Wpl[(C_ + k) * C_ + c];
  out2[(size_t)node * C_ + c] = s;
}

// ---------- phase 4: GraphNorm (stats over all B*N nodes) ----------
__global__ void k_gnstats(const float* __restrict__ out2, const float* __restrict__ alpha,
                          const float* __restrict__ gw, float* __restrict__ muA,
                          float* __restrict__ scl) {
  int c = blockIdx.x, tid = threadIdx.x;
  __shared__ float s1[256], s2[256];
  float a = 0.f, b = 0.f;
  for (int r = tid; r < NODES; r += 256) {
    float v = out2[(size_t)r * C_ + c];
    a += v; b = fmaf(v, v, b);
  }
  s1[tid] = a; s2[tid] = b; __syncthreads();
  for (int st = 128; st > 0; st >>= 1) {
    if (tid < st) { s1[tid] += s1[tid + st]; s2[tid] += s2[tid + st]; }
    __syncthreads();
  }
  if (tid == 0) {
    float mu  = s1[0] * (1.f / NODES);
    float al  = alpha[c];
    // var of (out - al*mu):  E[out^2] - (2a - a^2) mu^2
    float var = s2[0] * (1.f / NODES) - (2.f * al - al * al) * mu * mu;
    muA[c] = al * mu;
    scl[c] = gw[c] * rsqrtf(var + 1e-5f);
  }
}

__global__ void k_finish(const float* __restrict__ out2, const float* __restrict__ x,
                         const float* __restrict__ muA, const float* __restrict__ scl,
                         const float* __restrict__ gb, float* __restrict__ y) {
  int t = blockIdx.x * 256 + threadIdx.x;
  if (t >= NODES * C_) return;
  int c = t % C_;
  float v = scl[c] * (out2[t] - muA[c]) + gb[c];
  y[t] = fmaxf(v, 0.f) + x[t];
}

extern "C" void kernel_launch(void* const* d_in, const int* in_sizes, int n_in,
                              void* d_out, int out_size, void* d_ws, size_t ws_size,
                              hipStream_t stream) {
  (void)in_sizes; (void)n_in; (void)out_size; (void)ws_size;
  const float* x     = (const float*)d_in[0];
  const float* edge  = (const float*)d_in[1];
  const float* We    = (const float*)d_in[2];
  const float* be    = (const float*)d_in[3];
  const float* Wpre  = (const float*)d_in[4];
  const float* bpre  = (const float*)d_in[5];
  const float* Wpost = (const float*)d_in[6];
  const float* bpost = (const float*)d_in[7];
  const float* Wlin  = (const float*)d_in[8];
  const float* blin  = (const float*)d_in[9];
  const float* gw    = (const float*)d_in[10];
  const float* gb    = (const float*)d_in[11];
  const float* ga    = (const float*)d_in[12];
  float* y = (float*)d_out;

  char* ws = (char*)d_ws;
  unsigned short* wfold = (unsigned short*)(ws + WFOLD_OFF);
  float* bfold = (float*)(ws + BFOLD_OFF);
  float* Pi    = (float*)(ws + PI_OFF);
  float* Pj    = (float*)(ws + PJ_OFF);
  float* Wpl   = (float*)(ws + WPL_OFF);
  float* bpl   = (float*)(ws + BPL_OFF);
  float* part  = (float*)(ws + PART_OFF);
  float* agg   = (float*)(ws + AGG_OFF);
  float* out2  = (float*)(ws + OUT2_OFF);
  float* muA   = (float*)(ws + MUA_OFF);
  float* scl   = (float*)(ws + SCL_OFF);

  k_fold_edge<<<(C_ * C_ + C_ + 255) / 256, 256, 0, stream>>>(We, be, Wpre, bpre, wfold, bfold);
  k_pre_xproj<<<(2 * NODES * C_ + 255) / 256, 256, 0, stream>>>(x, Wpre, Pi, Pj);
  k_fold_post<<<(5 * C_ * C_ + C_ + 255) / 256, 256, 0, stream>>>(Wpost, bpost, Wlin, blin, Wpl, bpl);

  k_edge_pna<<<B_ * (N_ / 16) * JS, 256, 0, stream>>>(edge, wfold, bfold, Pi, Pj, part);

  k_combine<<<(NODES * C_ + 255) / 256, 256, 0, stream>>>(part, agg);
  k_post<<<(NODES * C_ + 255) / 256, 256, 0, stream>>>(x, agg, Wpl, bpl, out2);
  k_gnstats<<<C_, 256, 0, stream>>>(out2, ga, gw, muA, scl);
  k_finish<<<(NODES * C_ + 255) / 256, 256, 0, stream>>>(out2, x, muA, scl, gb, y);
}
